// PatchedDeepseekV2MoE_14645838479470
// MI455X (gfx1250) — compile-verified, matrix operand order checked by reading, not run on hardware
//
#include <hip/hip_runtime.h>

#define T_TOK 2048
#define HID   2048
#define FFN_  1408
#define NE    16
#define TOPK  4
#define SFFN_ 2816
#define CAP   (T_TOK * TOPK + 32 * NE)   // padded slot capacity = 8704
#define KSTEP 32
#define MBLK  32
#define NBLK  128
#define APAD  8
#define NTILES (NBLK / 16)               // 8 n-tiles of 16 per block
#define ALD   (KSTEP + APAD)

typedef __attribute__((ext_vector_type(16))) __bf16 v16bf;
typedef __attribute__((ext_vector_type(8)))  __bf16 v8bf;
typedef __attribute__((ext_vector_type(8)))  float  v8f;

// A tile: row-major As[m][ALD]; a lane's fragment is two contiguous
// 16-byte runs: k in [hi*8, hi*8+8) and [16+hi*8, 16+hi*8+8).
__device__ __forceinline__ v16bf frag_a(const __bf16* As, int m0, int lane) {
  const int m = lane & 15, hi = lane >> 4;
  const __bf16* p = As + (size_t)(m0 + m) * ALD + hi * 8;
  const v8bf lo = *reinterpret_cast<const v8bf*>(p);        // ds_load_b128
  const v8bf up = *reinterpret_cast<const v8bf*>(p + 16);   // ds_load_b128
  return __builtin_shufflevector(lo, up, 0, 1, 2, 3, 4, 5, 6, 7,
                                 8, 9, 10, 11, 12, 13, 14, 15);
}

// B tile stored in LDS pre-swizzled into fragment order:
// row = ((nt*2 + hi) << 4) + n_local, each row = one lane's 16 bf16 (32 B).
__device__ __forceinline__ v16bf frag_b(const __bf16* Bf, int nt, int lane) {
  const int row = (((nt << 1) + (lane >> 4)) << 4) + (lane & 15);
  return *reinterpret_cast<const v16bf*>(Bf + ((size_t)row << 4)); // 2x ds_load_b128
}

__device__ __forceinline__ v8bf pack8(const float4 a, const float4 b) {
  v8bf p;
  p[0] = (__bf16)a.x; p[1] = (__bf16)a.y; p[2] = (__bf16)a.z; p[3] = (__bf16)a.w;
  p[4] = (__bf16)b.x; p[5] = (__bf16)b.y; p[6] = (__bf16)b.z; p[7] = (__bf16)b.w;
  return p;
}

// Destination offset (elements) in the swizzled B buffer for a staging thread
// owning column n (=b_n) and an 8-wide k-run starting at b_k (b_k in {0,8,16,24}).
__device__ __forceinline__ int b_dst_off(int b_n, int b_k) {
  const int nt = b_n >> 4, n_local = b_n & 15;
  const int hi = (b_k >> 3) & 1;
  const int voff = (b_k & 16) >> 1;               // element 0 or 8
  return (((((nt << 1) + hi) << 4) + n_local) << 4) + voff;
}

// ---------------------------------------------------------------- small utils
__global__ __launch_bounds__(256) void init_kernel(int* slot_token, float* slot_w,
                                                   int* counts, int* cursors) {
  const int i = blockIdx.x * 256 + threadIdx.x;
  if (i < CAP) { slot_token[i] = -1; slot_w[i] = 0.f; }
  if (i < NE)  { counts[i] = 0; cursors[i] = 0; }
}

__global__ __launch_bounds__(256) void convert_kernel(const float* __restrict__ x,
                                                      __bf16* __restrict__ xb) {
  const size_t i = ((size_t)blockIdx.x * 256 + threadIdx.x) * 8;
  if (i < (size_t)T_TOK * HID) {
    const float4 v0 = *reinterpret_cast<const float4*>(x + i);
    const float4 v1 = *reinterpret_cast<const float4*>(x + i + 4);
    *reinterpret_cast<v8bf*>(xb + i) = pack8(v0, v1);
  }
}

// Router: softmax over 16 experts, top-4 (not renormalized), count tokens/expert.
__global__ __launch_bounds__(128) void gate_kernel(const float* __restrict__ x,
                                                   const float* __restrict__ gw,
                                                   int* __restrict__ topk_idx,
                                                   float* __restrict__ topk_w,
                                                   int* __restrict__ counts) {
  const int t = blockIdx.x;
  const int tid = threadIdx.x;
  const int e = tid >> 3, j = tid & 7;
  const float4* xr = reinterpret_cast<const float4*>(x + (size_t)t * HID);
  const float4* wr = reinterpret_cast<const float4*>(gw + (size_t)e * HID);
  float s = 0.f;
  for (int i = j; i < HID / 4; i += 8) {
    const float4 a = xr[i], b = wr[i];
    s += a.x * b.x + a.y * b.y + a.z * b.z + a.w * b.w;
  }
  __shared__ float red[128];
  __shared__ float sc[NE];
  red[tid] = s;
  __syncthreads();
  if (j == 0) {
    float v = 0.f;
#pragma unroll
    for (int q = 0; q < 8; ++q) v += red[e * 8 + q];
    sc[e] = v;
  }
  __syncthreads();
  if (tid == 0) {
    float p[NE];
    float mx = sc[0];
    for (int q = 1; q < NE; ++q) mx = fmaxf(mx, sc[q]);
    float se = 0.f;
    for (int q = 0; q < NE; ++q) { p[q] = __expf(sc[q] - mx); se += p[q]; }
    const float inv = 1.f / se;
    for (int q = 0; q < NE; ++q) p[q] *= inv;
    for (int k = 0; k < TOPK; ++k) {
      int am = 0; float bv = p[0];
      for (int q = 1; q < NE; ++q) if (p[q] > bv) { bv = p[q]; am = q; }
      topk_idx[t * TOPK + k] = am;
      topk_w[t * TOPK + k]   = bv;
      atomicAdd(&counts[am], 1);
      p[am] = -1.f;
    }
  }
}

// Exclusive prefix over expert counts, padded to MBLK so GEMM tiles never straddle experts.
__global__ void scan_kernel(const int* __restrict__ counts, int* __restrict__ offsets) {
  if (threadIdx.x == 0) {
    int acc = 0;
    for (int e = 0; e < NE; ++e) {
      offsets[e] = acc;
      acc += ((counts[e] + MBLK - 1) / MBLK) * MBLK;
    }
    offsets[NE] = acc;
  }
}

__global__ __launch_bounds__(256) void fill_kernel(const int* __restrict__ topk_idx,
                                                   const float* __restrict__ topk_w,
                                                   const int* __restrict__ offsets,
                                                   int* __restrict__ cursors,
                                                   int* __restrict__ slot_token,
                                                   float* __restrict__ slot_w) {
  const int t = blockIdx.x * 256 + threadIdx.x;
  if (t >= T_TOK) return;
#pragma unroll
  for (int k = 0; k < TOPK; ++k) {
    const int e = topk_idx[t * TOPK + k];
    const int pos = atomicAdd(&cursors[e], 1);
    const int s = offsets[e] + pos;
    slot_token[s] = t;
    slot_w[s]     = topk_w[t * TOPK + k];
  }
}

// ------------------- fused GLU up-projection: act = silu(x@Wg^T) * (x@Wu^T) --
// offsets==nullptr -> shared-expert path (identity token map, single weight set)
__global__ __launch_bounds__(256) void glu_up_kernel(
    const __bf16* __restrict__ xb, const float* __restrict__ W,
    const int* __restrict__ offsets, const int* __restrict__ slot_token,
    __bf16* __restrict__ actout, int Fd) {
  const int slot0 = blockIdx.x * MBLK;
  const float* Wg = W;
  if (offsets) {
    if (slot0 >= offsets[NE]) return;
    int e = 0;
    while (slot0 >= offsets[e + 1]) ++e;
    Wg = W + (size_t)e * 2 * (size_t)Fd * HID;
  }
  const int f0 = blockIdx.y * NBLK;

  __shared__ __align__(32) __bf16 As[2][MBLK][ALD];        // 2 x 2.5 KB
  __shared__ __align__(32) __bf16 Bg[2][NTILES * 32 * 16]; // 2 x 8 KB
  __shared__ __align__(32) __bf16 Bu[2][NTILES * 32 * 16]; // 2 x 8 KB

  const int tid   = threadIdx.x;
  const int a_row = tid >> 3, a_col = (tid & 7) * 4;
  const int b_n   = tid >> 1, b_k   = (tid & 1) * 16;      // 16 consecutive k
  const int tokenA = slot_token ? slot_token[slot0 + a_row] : (slot0 + a_row);
  const __bf16* aSrc = (tokenA >= 0) ? (xb + (size_t)tokenA * HID + a_col) : nullptr;
  const float* gRow = Wg + (size_t)(f0 + b_n) * HID;
  const float* uRow = Wg + (size_t)(Fd + f0 + b_n) * HID;
  const int dst0 = b_dst_off(b_n, b_k);
  const int dst1 = b_dst_off(b_n, b_k + 8);

  const int wave = tid >> 5, lane = tid & 31;              // wave = n-tile
  v8f cg0 = {}, cg1 = {}, cu0 = {}, cu1 = {};

  auto stage = [&](int p, int k0) {
    if (aSrc) {
      *reinterpret_cast<uint2*>(&As[p][a_row][a_col]) =
          *reinterpret_cast<const uint2*>(aSrc + k0);
    } else {
      As[p][a_row][a_col]     = (__bf16)0.f;  As[p][a_row][a_col + 1] = (__bf16)0.f;
      As[p][a_row][a_col + 2] = (__bf16)0.f;  As[p][a_row][a_col + 3] = (__bf16)0.f;
    }
    const float4 g0 = *reinterpret_cast<const float4*>(gRow + k0 + b_k);
    const float4 g1 = *reinterpret_cast<const float4*>(gRow + k0 + b_k + 4);
    const float4 g2 = *reinterpret_cast<const float4*>(gRow + k0 + b_k + 8);
    const float4 g3 = *reinterpret_cast<const float4*>(gRow + k0 + b_k + 12);
    const float4 u0 = *reinterpret_cast<const float4*>(uRow + k0 + b_k);
    const float4 u1 = *reinterpret_cast<const float4*>(uRow + k0 + b_k + 4);
    const float4 u2 = *reinterpret_cast<const float4*>(uRow + k0 + b_k + 8);
    const float4 u3 = *reinterpret_cast<const float4*>(uRow + k0 + b_k + 12);
    *reinterpret_cast<v8bf*>(&Bg[p][dst0]) = pack8(g0, g1);   // ds_store_b128
    *reinterpret_cast<v8bf*>(&Bg[p][dst1]) = pack8(g2, g3);
    *reinterpret_cast<v8bf*>(&Bu[p][dst0]) = pack8(u0, u1);
    *reinterpret_cast<v8bf*>(&Bu[p][dst1]) = pack8(u2, u3);
  };

  stage(0, 0);
  __syncthreads();
  int p = 0;
  for (int k0 = 0; k0 < HID; k0 += KSTEP) {
    if (k0 + KSTEP < HID) {
      stage(p ^ 1, k0 + KSTEP);
      __builtin_prefetch(gRow + k0 + 2 * KSTEP + b_k, 0, 1);
      __builtin_prefetch(uRow + k0 + 2 * KSTEP + b_k, 0, 1);
    }
    const v16bf a0 = frag_a(&As[p][0][0], 0, lane);
    const v16bf a1 = frag_a(&As[p][0][0], 16, lane);
    const v16bf bg = frag_b(Bg[p], wave, lane);
    const v16bf bu = frag_b(Bu[p], wave, lane);
    cg0 = __builtin_amdgcn_wmma_f32_16x16x32_bf16(false, a0, false, bg, (short)0, cg0, false, false);
    cg1 = __builtin_amdgcn_wmma_f32_16x16x32_bf16(false, a1, false, bg, (short)0, cg1, false, false);
    cu0 = __builtin_amdgcn_wmma_f32_16x16x32_bf16(false, a0, false, bu, (short)0, cu0, false, false);
    cu1 = __builtin_amdgcn_wmma_f32_16x16x32_bf16(false, a1, false, bu, (short)0, cu1, false, false);
    __syncthreads();
    p ^= 1;
  }

  const int n = lane & 15, hi = lane >> 4;
  const int fcol = f0 + wave * 16 + n;
#pragma unroll
  for (int mt = 0; mt < 2; ++mt) {
    const v8f g8 = mt ? cg1 : cg0;
    const v8f u8 = mt ? cu1 : cu0;
#pragma unroll
    for (int r = 0; r < 8; ++r) {
      const int slot = slot0 + mt * 16 + r + 8 * hi;
      const float g = g8[r], u = u8[r];
      const float s = g / (1.f + __expf(-g));          // silu(g) * u
      actout[(size_t)slot * Fd + fcol] = (__bf16)(s * u);
    }
  }
}

// -------------------------- down projection: y (+)= w_slot * (act @ Wdown^T) --
// offsets==nullptr -> shared path (plain store, initializes y). Routed path is
// launched once per expert sequentially -> unique y rows, no atomics needed.
__global__ __launch_bounds__(256) void down_kernel(
    const __bf16* __restrict__ actin, const float* __restrict__ W,
    const int* __restrict__ offsets, const int* __restrict__ slot_token,
    const float* __restrict__ slot_w, float* __restrict__ y, int Kd, int e) {
  int slot0;
  const float* We = W;
  if (offsets) {
    const int base = offsets[e], end = offsets[e + 1];
    slot0 = base + blockIdx.x * MBLK;
    if (slot0 >= end) return;
    We = W + (size_t)e * HID * (size_t)Kd;
  } else {
    slot0 = blockIdx.x * MBLK;
  }
  const int h0 = blockIdx.y * NBLK;

  __shared__ __align__(32) __bf16 As[2][MBLK][ALD];        // 2 x 2.5 KB
  __shared__ __align__(32) __bf16 Bf[2][NTILES * 32 * 16]; // 2 x 8 KB

  const int tid   = threadIdx.x;
  const int a_row = tid >> 3, a_col = (tid & 7) * 4;
  const int b_n   = tid >> 1, b_k   = (tid & 1) * 16;
  const __bf16* aSrc = actin + (size_t)(slot0 + a_row) * Kd + a_col;
  const float*  wRow = We + (size_t)(h0 + b_n) * Kd;
  const int dst0 = b_dst_off(b_n, b_k);
  const int dst1 = b_dst_off(b_n, b_k + 8);

  const int wave = tid >> 5, lane = tid & 31;
  v8f c0 = {}, c1 = {};

  auto stage = [&](int p, int k0) {
    *reinterpret_cast<uint2*>(&As[p][a_row][a_col]) =
        *reinterpret_cast<const uint2*>(aSrc + k0);
    const float4 w0 = *reinterpret_cast<const float4*>(wRow + k0 + b_k);
    const float4 w1 = *reinterpret_cast<const float4*>(wRow + k0 + b_k + 4);
    const float4 w2 = *reinterpret_cast<const float4*>(wRow + k0 + b_k + 8);
    const float4 w3 = *reinterpret_cast<const float4*>(wRow + k0 + b_k + 12);
    *reinterpret_cast<v8bf*>(&Bf[p][dst0]) = pack8(w0, w1);   // ds_store_b128
    *reinterpret_cast<v8bf*>(&Bf[p][dst1]) = pack8(w2, w3);
  };

  stage(0, 0);
  __syncthreads();
  int p = 0;
  for (int k0 = 0; k0 < Kd; k0 += KSTEP) {
    if (k0 + KSTEP < Kd) {
      stage(p ^ 1, k0 + KSTEP);
      __builtin_prefetch(wRow + k0 + 2 * KSTEP + b_k, 0, 1);
    }
    const v16bf a0 = frag_a(&As[p][0][0], 0, lane);
    const v16bf a1 = frag_a(&As[p][0][0], 16, lane);
    const v16bf b = frag_b(Bf[p], wave, lane);
    c0 = __builtin_amdgcn_wmma_f32_16x16x32_bf16(false, a0, false, b, (short)0, c0, false, false);
    c1 = __builtin_amdgcn_wmma_f32_16x16x32_bf16(false, a1, false, b, (short)0, c1, false, false);
    __syncthreads();
    p ^= 1;
  }

  const int n = lane & 15, hi = lane >> 4;
  const int col = h0 + wave * 16 + n;
#pragma unroll
  for (int mt = 0; mt < 2; ++mt) {
    const v8f cc = mt ? c1 : c0;
#pragma unroll
    for (int r = 0; r < 8; ++r) {
      const int slot = slot0 + mt * 16 + r + 8 * hi;
      if (offsets) {
        const int token = slot_token[slot];
        if (token >= 0)
          y[(size_t)token * HID + col] += slot_w[slot] * cc[r];
      } else {
        y[(size_t)slot * HID + col] = cc[r];
      }
    }
  }
}

// ---------------------------------------------------------------------- host
extern "C" void kernel_launch(void* const* d_in, const int* in_sizes, int n_in,
                              void* d_out, int out_size, void* d_ws, size_t ws_size,
                              hipStream_t stream) {
  (void)in_sizes; (void)n_in; (void)out_size; (void)ws_size;
  const float* x   = (const float*)d_in[0];   // [T,H]
  const float* gw  = (const float*)d_in[1];   // [E,H]
  const float* gup = (const float*)d_in[2];   // [E,2F,H]
  const float* dwn = (const float*)d_in[3];   // [E,H,F]
  const float* sgu = (const float*)d_in[4];   // [2*SF,H]
  const float* sdn = (const float*)d_in[5];   // [H,SF]
  float* y = (float*)d_out;                   // [T,H]

  char* ws = (char*)d_ws;
  size_t off = 0;
  auto alloc = [&](size_t bytes) -> void* {
    void* p = ws + off;
    off = (off + bytes + 255) & ~(size_t)255;
    return p;
  };
  __bf16* xb        = (__bf16*)alloc((size_t)T_TOK * HID * 2);
  int*    topk_idx  = (int*)   alloc((size_t)T_TOK * TOPK * 4);
  float*  topk_w    = (float*) alloc((size_t)T_TOK * TOPK * 4);
  int*    counts    = (int*)   alloc(NE * 4);
  int*    offsets   = (int*)   alloc((NE + 1) * 4);
  int*    cursors   = (int*)   alloc(NE * 4);
  int*    slot_tok  = (int*)   alloc((size_t)CAP * 4);
  float*  slot_w    = (float*) alloc((size_t)CAP * 4);
  __bf16* act       = (__bf16*)alloc((size_t)CAP * FFN_ * 2);
  __bf16* act_sh    = (__bf16*)alloc((size_t)T_TOK * SFFN_ * 2);

  // 1) routing + dispatch tables
  init_kernel<<<(CAP + 255) / 256, 256, 0, stream>>>(slot_tok, slot_w, counts, cursors);
  convert_kernel<<<((size_t)T_TOK * HID / 8 + 255) / 256, 256, 0, stream>>>(x, xb);
  gate_kernel<<<T_TOK, 128, 0, stream>>>(x, gw, topk_idx, topk_w, counts);
  scan_kernel<<<1, 32, 0, stream>>>(counts, offsets);
  fill_kernel<<<T_TOK / 256, 256, 0, stream>>>(topk_idx, topk_w, offsets, cursors,
                                               slot_tok, slot_w);

  // 2) shared expert (down pass initializes y with plain stores)
  dim3 gsu(T_TOK / MBLK, SFFN_ / NBLK);    // 64 x 22
  glu_up_kernel<<<gsu, 256, 0, stream>>>(xb, sgu, nullptr, nullptr, act_sh, SFFN_);
  dim3 gsd(T_TOK / MBLK, HID / NBLK);      // 64 x 16
  down_kernel<<<gsd, 256, 0, stream>>>(act_sh, sdn, nullptr, nullptr, nullptr, y,
                                       SFFN_, 0);

  // 3) routed experts over compacted slots
  dim3 gru(CAP / MBLK, FFN_ / NBLK);       // 272 x 11 (early-exit past total)
  glu_up_kernel<<<gru, 256, 0, stream>>>(xb, gup, offsets, slot_tok, act, FFN_);
  dim3 grd(T_TOK / MBLK, HID / NBLK);      // worst-case per-expert tiles
  for (int e = 0; e < NE; ++e)
    down_kernel<<<grd, 256, 0, stream>>>(act, dwn, offsets, slot_tok, slot_w, y,
                                         FFN_, e);
}
